// PREFIX_78116865180378
// MI455X (gfx1250) — matrix operand order repaired and checked by reading, not compiled, not run on hardware
//
#include <hip/hip_runtime.h>
#include <hip/hip_bf16.h>
#include <math.h>

#define TOPK 8
#define SEQ 128
#define BSZ 64
#define D 256
#define VOCAB 5000
#define NTOK (BSZ * SEQ * TOPK)   // 65536 rows of the gathered prefix matrix

typedef __attribute__((ext_vector_type(16))) _Float16 v16h;
typedef __attribute__((ext_vector_type(8)))  _Float16 v8h;
typedef __attribute__((ext_vector_type(8)))  float    v8f;

// ---------------------------------------------------------------------------
// WMMA fragment loaders (this code's contiguous k order: lane half h holds k0+16h..k0+16h+15, wave32)
// ---------------------------------------------------------------------------

// A-matrix 16x32 f16, row-major source [row][k], ld = row stride (halfs).
// lanes 0-15: M=lane, K = k0..k0+15
// lanes16-31: M=lane-16, K = k0+16..k0+31
__device__ __forceinline__ v16h load_a_frag(const _Float16* base, int row0,
                                            int ld, int k0, int lane) {
  const int sel = (lane & 16) ? 16 : 0;
  const int r   = row0 + (lane & 15);
  const _Float16* p = base + (size_t)r * ld + k0 + sel;
  v8h lo = *(const v8h*)(p);
  v8h hi = *(const v8h*)(p + 8);
  v16h out;
#pragma unroll
  for (int i = 0; i < 8; ++i) { out[i] = lo[i]; out[8 + i] = hi[i]; }
  return out;
}

// B-matrix 32x16 f16 from a K-contiguous layout: src[col][k] (ld halfs/col).
// lanes 0-15: N=col0+lane,    K=k0..k0+15   (16 contiguous halfs)
// lanes16-31: N=col0+lane-16, K=k0+16..k0+31
__device__ __forceinline__ v16h load_b_frag(const _Float16* base, int col0,
                                            int ld, int k0, int lane) {
  const int n  = col0 + (lane & 15);
  const int kk = k0 + ((lane & 16) ? 16 : 0);
  return *(const v16h*)(base + (size_t)n * ld + kk);
}

// B-fragment over key tokens with clamping (for banded attention edges).
__device__ __forceinline__ v16h load_b_frag_tok(const _Float16* Kmat,
                                                int batch_base, int tok_off,
                                                int k0, int lane) {
  int t = tok_off + (lane & 15);
  t = t < 0 ? 0 : (t > (SEQ * TOPK - 1) ? (SEQ * TOPK - 1) : t);
  const int kk = k0 + ((lane & 16) ? 16 : 0);
  return *(const v16h*)(Kmat + (size_t)(batch_base + t) * D + kk);
}

// ---------------------------------------------------------------------------
// Kernel 1: full softmax over vocab + top-8 + renormalizing softmax.
// One 256-thread block per (b,s) row; vocab row staged in LDS once.
// ---------------------------------------------------------------------------
__global__ __launch_bounds__(256) void topk_kernel(
    const float* __restrict__ na_pred, int* __restrict__ idx_out,
    float* __restrict__ pP_out, float* __restrict__ na_score_zero,
    float* __restrict__ idx_f_out) {
  __shared__ float x[VOCAB];
  __shared__ float rv[256];
  __shared__ int   ri[256];
  __shared__ float tv[TOPK];
  __shared__ int   ti[TOPK];
  __shared__ float s_max, s_sum;

  const int r   = blockIdx.x;            // row in [0, BSZ*SEQ)
  const int tid = threadIdx.x;
  const float* row = na_pred + (size_t)r * VOCAB;

  float lm = -INFINITY;
  for (int i = tid; i < VOCAB; i += 256) {
    float v = row[i];
    x[i] = v;
    lm = fmaxf(lm, v);
  }
  rv[tid] = lm;
  __syncthreads();
  for (int s = 128; s > 0; s >>= 1) {
    if (tid < s) rv[tid] = fmaxf(rv[tid], rv[tid + s]);
    __syncthreads();
  }
  if (tid == 0) s_max = rv[0];
  __syncthreads();

  float ls = 0.f;
  const float mx = s_max;
  for (int i = tid; i < VOCAB; i += 256) ls += expf(x[i] - mx);
  rv[tid] = ls;
  __syncthreads();
  for (int s = 128; s > 0; s >>= 1) {
    if (tid < s) rv[tid] += rv[tid + s];
    __syncthreads();
  }
  if (tid == 0) s_sum = rv[0];
  __syncthreads();

  // 8 iterative argmax passes over the LDS copy.
  for (int t = 0; t < TOPK; ++t) {
    float bm = -INFINITY; int bi = 0;
    for (int i = tid; i < VOCAB; i += 256) {
      float v = x[i];
      if (v > bm || (v == bm && i < bi)) { bm = v; bi = i; }
    }
    rv[tid] = bm; ri[tid] = bi;
    __syncthreads();
    for (int s = 128; s > 0; s >>= 1) {
      if (tid < s) {
        float ov = rv[tid + s]; int oi = ri[tid + s];
        if (ov > rv[tid] || (ov == rv[tid] && oi < ri[tid])) {
          rv[tid] = ov; ri[tid] = oi;
        }
      }
      __syncthreads();
    }
    if (tid == 0) { tv[t] = rv[0]; ti[t] = ri[0]; x[ri[0]] = -INFINITY; }
    __syncthreads();
  }

  if (tid == 0) {
    // p = softmax(logits)[topk]; then softmax again over the 8 probs.
    float p[TOPK];
    const float z = s_sum;
    float m2 = -INFINITY;
    for (int t = 0; t < TOPK; ++t) { p[t] = expf(tv[t] - mx) / z; m2 = fmaxf(m2, p[t]); }
    float s2 = 0.f;
    for (int t = 0; t < TOPK; ++t) { p[t] = expf(p[t] - m2); s2 += p[t]; }
    for (int t = 0; t < TOPK; ++t) {
      const int o = r * TOPK + t;
      pP_out[o]        = p[t] / s2;
      idx_out[o]       = ti[t];
      idx_f_out[o]     = (float)ti[t];
      na_score_zero[o] = 0.f;   // re-zeroed every launch (graph replay safe)
    }
  }
}

// ---------------------------------------------------------------------------
// Kernel 2: gather prefix embeddings, scale by sqrt(D)=16, convert to f16.
// ---------------------------------------------------------------------------
__global__ __launch_bounds__(256) void gather_kernel(
    const int* __restrict__ idx, const float* __restrict__ emb,
    _Float16* __restrict__ Eh) {
  const int t = blockIdx.x;         // token row in [0, NTOK)
  const int d = threadIdx.x;        // dim
  const int v = idx[t];
  Eh[(size_t)t * D + d] = (_Float16)(emb[(size_t)v * D + d] * 16.0f);
}

// Kernel 3: transpose Wq/Wk to f16 [n][k] so B-fragments are K-contiguous.
__global__ __launch_bounds__(256) void wtrans_kernel(
    const float* __restrict__ Wq, const float* __restrict__ Wk,
    _Float16* __restrict__ WT) {  // WT[p][n*D + k]
  const int n = blockIdx.x, p = blockIdx.y, k = threadIdx.x;
  const float* W = p ? Wk : Wq;
  WT[(size_t)p * D * D + (size_t)n * D + k] = (_Float16)W[(size_t)k * D + n];
}

// ---------------------------------------------------------------------------
// Kernel 4: projections Q = Eh@Wq + bq, K = Eh@Wk + bk via f16 WMMA.
// 8 waves/block, each wave owns a 16-row M-tile and sweeps all 256 N cols.
// ---------------------------------------------------------------------------
__global__ __launch_bounds__(256) void proj_kernel(
    const _Float16* __restrict__ Eh, const _Float16* __restrict__ WT,
    const float* __restrict__ bq, const float* __restrict__ bk,
    _Float16* __restrict__ Qh, _Float16* __restrict__ Kh) {
  const int lane = threadIdx.x & 31;
  const int wave = threadIdx.x >> 5;
  const int p    = blockIdx.y;
  const int m0   = (blockIdx.x * 8 + wave) * 16;

  const _Float16* W   = WT + (size_t)p * D * D;
  const float*    bia = p ? bk : bq;
  _Float16*       Out = p ? Kh : Qh;

#pragma unroll
  for (int ng = 0; ng < 4; ++ng) {           // 4 groups of 4 N-tiles = 256 cols
    const int n0b = ng * 64;
    v8f acc[4];
#pragma unroll
    for (int j = 0; j < 4; ++j) acc[j] = (v8f)(0.f);

#pragma unroll
    for (int k0 = 0; k0 < D; k0 += 32) {
      const v16h a = load_a_frag(Eh, m0, D, k0, lane);
#pragma unroll
      for (int j = 0; j < 4; ++j) {
        const v16h b = load_b_frag(W, n0b + j * 16, D, k0, lane);
        acc[j] = __builtin_amdgcn_wmma_f32_16x16x32_f16(
            false, a, false, b, (short)0, acc[j], false, false);
      }
    }

    // C layout: row = v + 8*(lane>=16), col = lane&15; add bias, store f16.
    const int rbase = m0 + ((lane & 16) ? 8 : 0);
#pragma unroll
    for (int j = 0; j < 4; ++j) {
      const int n  = n0b + j * 16 + (lane & 15);
      const float bn = bia[n];
#pragma unroll
      for (int v = 0; v < 8; ++v)
        Out[(size_t)(rbase + v) * D + n] = (_Float16)(acc[j][v] + bn);
    }
  }
}

// ---------------------------------------------------------------------------
// Kernel 5: banded attention. One wave handles the query-chunk pair (2c,2c+1)
// of one batch: 16x32 score tile via WMMA (keys = chunks 2c-1..2c+2), row
// softmax over the 16 allowed keys in LDS, atomic column-sum into na_score.
// ---------------------------------------------------------------------------
__global__ __launch_bounds__(256) void attn_kernel(
    const _Float16* __restrict__ Qh, const _Float16* __restrict__ Kh,
    float* __restrict__ na_score) {
  __shared__ float sc[8][16][32];   // per-wave private 16x32 score tiles

  const int lane = threadIdx.x & 31;
  const int wave = threadIdx.x >> 5;
  const int id   = blockIdx.x * 8 + wave;   // [0, 64*64)
  const int b    = id >> 6;
  const int c    = id & 63;                 // query-chunk pair index

  const int bbase = b * SEQ * TOPK;
  const int qrow0 = bbase + c * 16;
  const int kb0   = c * 16 - 8;   // start token of chunk 2c-1 (may be <0)
  const int kb1   = c * 16 + 8;   // start token of chunk 2c+1 (may run past end)

  v8f acc0 = (v8f)(0.f), acc1 = (v8f)(0.f);
#pragma unroll
  for (int k0 = 0; k0 < D; k0 += 32) {
    const v16h a  = load_a_frag(Qh, qrow0, D, k0, lane);
    const v16h b0 = load_b_frag_tok(Kh, bbase, kb0, k0, lane);
    acc0 = __builtin_amdgcn_wmma_f32_16x16x32_f16(false, a, false, b0,
                                                  (short)0, acc0, false, false);
    const v16h b1 = load_b_frag_tok(Kh, bbase, kb1, k0, lane);
    acc1 = __builtin_amdgcn_wmma_f32_16x16x32_f16(false, a, false, b1,
                                                  (short)0, acc1, false, false);
  }

  const int row = (lane & 16) ? 8 : 0;
  const int col = lane & 15;
#pragma unroll
  for (int v = 0; v < 8; ++v) {
    sc[wave][row + v][col]      = acc0[v] * 0.0625f;   // /sqrt(D)
    sc[wave][row + v][16 + col] = acc1[v] * 0.0625f;
  }
  __syncthreads();

  if (lane < 16) {
    const int r = lane;                 // local query row
    const bool lower = (r < 8);
    // allowed LDS columns: lower rows -> {0..7} (chunk 2c-1), {16..23} (2c+1)
    //                      upper rows -> {8..15} (chunk 2c),  {24..31} (2c+2)
    const int cA = lower ? 0 : 8, cB = lower ? 16 : 24;
    const bool vA = lower ? (c > 0) : true;
    const bool vB = lower ? true : (c < 63);
    const int chA = lower ? (2 * c - 1) : (2 * c);
    const int chB = lower ? (2 * c + 1) : (2 * c + 2);

    float m = -INFINITY;
#pragma unroll
    for (int j = 0; j < 8; ++j) {
      if (vA) m = fmaxf(m, sc[wave][r][cA + j]);
      if (vB) m = fmaxf(m, sc[wave][r][cB + j]);
    }
    float s = 0.f;
#pragma unroll
    for (int j = 0; j < 8; ++j) {
      if (vA) s += expf(sc[wave][r][cA + j] - m);
      if (vB) s += expf(sc[wave][r][cB + j] - m);
    }
    const float inv = 1.0f / s;
#pragma unroll
    for (int j = 0; j < 8; ++j) {
      if (vA) atomicAdd(&na_score[bbase + chA * TOPK + j],
                        expf(sc[wave][r][cA + j] - m) * inv);
      if (vB) atomicAdd(&na_score[bbase + chB * TOPK + j],
                        expf(sc[wave][r][cB + j] - m) * inv);
    }
  }
}

// ---------------------------------------------------------------------------
// Kernel 6: prefix_score = L1norm(na_score * pP); scored_prefix = sum_k
//           prefix_score[k] * 16*emb[idx[k]].  One block per (b,s).
// ---------------------------------------------------------------------------
__global__ __launch_bounds__(256) void finalize_kernel(
    const float* __restrict__ na_score, const float* __restrict__ pP,
    const int* __restrict__ idx, const float* __restrict__ emb,
    float* __restrict__ out_emb, float* __restrict__ out_score) {
  __shared__ float ps[TOPK];
  const int r = blockIdx.x;           // (b,s) row in [0, BSZ*SEQ)
  const int tid = threadIdx.x;

  if (tid == 0) {
    float raw[TOPK]; float l1 = 0.f;
    for (int k = 0; k < TOPK; ++k) {
      raw[k] = na_score[r * TOPK + k] * pP[r * TOPK + k];
      l1 += fabsf(raw[k]);
    }
    l1 = fmaxf(l1, 1e-12f);
    for (int k = 0; k < TOPK; ++k) ps[k] = raw[k] / l1;
  }
  __syncthreads();
  if (tid < TOPK) out_score[r * TOPK + tid] = ps[tid];

  float s = 0.f;
#pragma unroll
  for (int k = 0; k < TOPK; ++k)
    s += ps[k] * emb[(size_t)idx[r * TOPK + k] * D + tid];
  out_emb[(size_t)r * D + tid] = s * 16.0f;
}

// ---------------------------------------------------------------------------
extern "C" void kernel_launch(void* const* d_in, const int* in_sizes, int n_in,
                              void* d_out, int out_size, void* d_ws, size_t ws_size,
                              hipStream_t stream) {
  (void)in_sizes; (void)n_in; (void)out_size; (void)ws_size;
  // inputs: bsz, seq_len, na_pred, emb_table, Wq, bq, Wk, bk
  const float* na_pred = (const float*)d_in[2];
  const float* emb     = (const float*)d_in[3];
  const float* Wq      = (const float*)d_in[4];
  const float* bq      = (const float*)d_in[5];
  const float* Wk      = (const float*)d_in[6];
  const float* bk      = (const float*)d_in[7];

  float* out = (float*)d_out;
  float* out_emb   = out;                               // (64,128,256)
  float* out_score = out + (size_t)BSZ * SEQ * D;       // (64,128,8)
  float* out_idx_f = out_score + (size_t)BSZ * SEQ * TOPK;

  // scratch layout (all offsets 256B aligned)
  char* w = (char*)d_ws;
  int*      idx_ws = (int*)(w + 0);                       // 256 KB
  float*    pP_ws  = (float*)(w + 262144);                // 256 KB
  float*    nas_ws = (float*)(w + 524288);                // 256 KB
  _Float16* Eh     = (_Float16*)(w + 786432);             // 32 MB
  _Float16* WT     = (_Float16*)(w + 786432 + 33554432);  // 256 KB
  _Float16* Qh     = (_Float16*)(w + 34603008);           // 32 MB
  _Float16* Kh     = (_Float16*)(w + 68157440);           // 32 MB

  topk_kernel<<<BSZ * SEQ, 256, 0, stream>>>(na_pred, idx_ws, pP_ws, nas_ws,
                                             out_idx_f);
  gather_kernel<<<NTOK, 256, 0, stream>>>(idx_ws, emb, Eh);
  wtrans_kernel<<<dim3(D, 2), 256, 0, stream>>>(Wq, Wk, WT);
  proj_kernel<<<dim3(NTOK / 128, 2), 256, 0, stream>>>(Eh, WT, bq, bk, Qh, Kh);
  attn_kernel<<<BSZ * 64 / 8, 256, 0, stream>>>(Qh, Kh, nas_ws);
  finalize_kernel<<<BSZ * SEQ, 256, 0, stream>>>(nas_ws, pP_ws, idx_ws, emb,
                                                 out_emb, out_score);
}
